// ContrastiveRankingLoss_32719060861502
// MI455X (gfx1250) — compile-verified, matrix operand order checked by reading, not run on hardware
//
#include <hip/hip_runtime.h>
#include <hip/hip_fp16.h>

typedef __attribute__((ext_vector_type(16))) _Float16 v16h;
typedef __attribute__((ext_vector_type(8)))  float    v8f;

#define B_ROWS   4096
#define N_ROWS   8192
#define DIMS     256
#define NDIMS    19
#define INV_T    (1.0f/0.07f)
#define LAMBDA_C 0.3f
#define AMB_TH   0.05f
#define LS       0.1f
#define EPS_N    1e-8f
#define NTILES   (N_ROWS/16)

// ---------------- workspace layout (bytes) ----------------
// [0x000000, 0x400000) : zh   — normalized z as f16, 8192*256*2
// [0x400000, 0x408000) : neg_lse f32[8192]
// [0x408000, 0x410000) : pieces  i32[8192]
// [0x410000, 0x411000) : hist    i32[1024]
// [0x411000, ...)      : scalars f32[0]=contrast_sum f32[1]=rank_sum u32[2]=rank_cnt

union Frag { v16h h; uint4 q[2]; };

__global__ void init_ws_kernel(unsigned int* hist_and_scalars) {
    for (int i = threadIdx.x; i < 1024 + 8; i += blockDim.x)
        hist_and_scalars[i] = 0u;
}

__global__ void pieces_hist_kernel(const int* __restrict__ pa, const int* __restrict__ pb,
                                   int* __restrict__ pieces, int* __restrict__ hist) {
    int i = blockIdx.x * blockDim.x + threadIdx.x;
    if (i < N_ROWS) {
        int p = (i < B_ROWS) ? pa[i] : pb[i - B_ROWS];
        pieces[i] = p;
        atomicAdd(&hist[p], 1);
    }
}

// one block (256 threads) per row: row L2-norm + f16 convert
__global__ void normalize_f16_kernel(const float* __restrict__ za, const float* __restrict__ zb,
                                     _Float16* __restrict__ zh) {
    __shared__ float red[256];
    int row = blockIdx.x;
    const float* src = (row < B_ROWS) ? (za + (size_t)row * DIMS)
                                      : (zb + (size_t)(row - B_ROWS) * DIMS);
    int t = threadIdx.x;
    float x = src[t];
    red[t] = x * x;
    __syncthreads();
    for (int s = 128; s > 0; s >>= 1) {
        if (t < s) red[t] += red[t + s];
        __syncthreads();
    }
    float scale = 1.0f / fmaxf(sqrtf(red[0]), EPS_N);
    zh[(size_t)row * DIMS + t] = (_Float16)(x * scale);
}

// issue this wave's quarter (2 KB) of the 8 KB B panel for tile `ct` into LDS `lb`
__device__ __forceinline__ void issue_panel(unsigned long long gz, int ct,
                                            unsigned int lb, unsigned int poff) {
    unsigned long long g = gz + (unsigned long long)ct * 8192ull + poff;
    unsigned int l = lb + poff;
#pragma unroll
    for (int q = 0; q < 4; ++q) {
        asm volatile("global_load_async_to_lds_b128 %0, %1, off"
                     :: "v"(l + q * 512), "v"(g + (unsigned long long)(q * 512))
                     : "memory");
    }
}

// Pass A: 4 waves/block = 64 rows; sweep all column tiles with the B panel
// staged in double-buffered LDS via async DMA.  Fixed-shift logsumexp
// (logits bounded by 1/T) -> one exp per element, branchless, no rescale.
__global__ __launch_bounds__(128) void infonce_neglse_kernel(
        const _Float16* __restrict__ zh, const int* __restrict__ pieces,
        float* __restrict__ neg_lse) {
    __shared__ __align__(16) char bpanel[2][8192];
    int tid  = threadIdx.x;
    int wave = tid >> 5;
    int lane = tid & 31;
    int half = lane >> 4;
    int ln   = lane & 15;
    int r0   = blockIdx.x * 64 + wave * 16;
    int k0   = half * 8;

    Frag a[8];
    const _Float16* arow = zh + (size_t)(r0 + ln) * DIMS;
#pragma unroll
    for (int ks = 0; ks < 8; ++ks) {
        a[ks].q[0] = *(const uint4*)(arow + ks * 32 + k0);
        a[ks].q[1] = *(const uint4*)(arow + ks * 32 + k0 + 16);
    }
    int pi[8];
#pragma unroll
    for (int v = 0; v < 8; ++v) pi[v] = pieces[r0 + 8 * half + v];

    float s[8];
#pragma unroll
    for (int v = 0; v < 8; ++v) s[v] = 0.0f;

    unsigned int poff = (unsigned int)(wave * 2048 + lane * 16);
    unsigned long long gz = (unsigned long long)(uintptr_t)zh;
    unsigned int lds0 = (unsigned int)(uintptr_t)&bpanel[0][0];
    unsigned int lds1 = (unsigned int)(uintptr_t)&bpanel[1][0];

    issue_panel(gz, 0, lds0, poff);                 // prime buffer 0

    for (int ct = 0; ct < NTILES; ++ct) {
        int nxt = (ct + 1) & (NTILES - 1);
        __syncthreads();                            // all done reading buf[(ct+1)&1]
        issue_panel(gz, nxt, ((ct + 1) & 1) ? lds1 : lds0, poff);
        asm volatile("s_wait_asynccnt 0x4" ::: "memory");  // panel ct landed (in-order)
        __syncthreads();                            // ... for every wave

        const char* bb = (ct & 1) ? (const char*)&bpanel[1][0]
                                  : (const char*)&bpanel[0][0];
        int c0 = ct * 16;
        v8f c = {};
#pragma unroll
        for (int ks = 0; ks < 8; ++ks) {
            Frag b;
            b.q[0] = *(const uint4*)(bb + ln * 512 + ks * 64 + k0 * 2);
            b.q[1] = *(const uint4*)(bb + ln * 512 + ks * 64 + k0 * 2 + 32);
            c = __builtin_amdgcn_wmma_f32_16x16x32_f16(
                    false, a[ks].h, false, b.h, (short)0, c, false, false);
        }
        int pj = pieces[c0 + ln];
#pragma unroll
        for (int v = 0; v < 8; ++v) {
            float arg = (pj != pi[v]) ? (c[v] * INV_T - INV_T) : -1e38f;
            s[v] += __expf(arg);                    // masked -> exp underflows to 0
        }
    }
#pragma unroll
    for (int v = 0; v < 8; ++v) {
        for (int off = 1; off < 16; off <<= 1)
            s[v] += __shfl_xor(s[v], off, 32);
        if (ln == 0)
            neg_lse[r0 + 8 * half + v] = (s[v] > 0.0f) ? (INV_T + __logf(s[v])) : -1e30f;
    }
}

// Pass B: same pipeline; positives of valid anchors accumulate
// log1p(exp(neg_lse[i] - sim_ij)) (branchy: positives are ~0.1% of elements).
__global__ __launch_bounds__(128) void infonce_pos_kernel(
        const _Float16* __restrict__ zh, const int* __restrict__ pieces,
        const float* __restrict__ neg_lse, const int* __restrict__ hist,
        float* __restrict__ contrast_sum) {
    __shared__ __align__(16) char bpanel[2][8192];
    int tid  = threadIdx.x;
    int wave = tid >> 5;
    int lane = tid & 31;
    int half = lane >> 4;
    int ln   = lane & 15;
    int r0   = blockIdx.x * 64 + wave * 16;
    int k0   = half * 8;

    Frag a[8];
    const _Float16* arow = zh + (size_t)(r0 + ln) * DIMS;
#pragma unroll
    for (int ks = 0; ks < 8; ++ks) {
        a[ks].q[0] = *(const uint4*)(arow + ks * 32 + k0);
        a[ks].q[1] = *(const uint4*)(arow + ks * 32 + k0 + 16);
    }
    int pi[8]; float nl[8]; int vld[8];
#pragma unroll
    for (int v = 0; v < 8; ++v) {
        int i   = r0 + 8 * half + v;
        pi[v]   = pieces[i];
        nl[v]   = neg_lse[i];
        int cnt = hist[pi[v]];
        vld[v]  = (cnt >= 2) && (cnt <= N_ROWS - 1);
    }

    unsigned int poff = (unsigned int)(wave * 2048 + lane * 16);
    unsigned long long gz = (unsigned long long)(uintptr_t)zh;
    unsigned int lds0 = (unsigned int)(uintptr_t)&bpanel[0][0];
    unsigned int lds1 = (unsigned int)(uintptr_t)&bpanel[1][0];

    issue_panel(gz, 0, lds0, poff);

    float acc = 0.0f;
    for (int ct = 0; ct < NTILES; ++ct) {
        int nxt = (ct + 1) & (NTILES - 1);
        __syncthreads();
        issue_panel(gz, nxt, ((ct + 1) & 1) ? lds1 : lds0, poff);
        asm volatile("s_wait_asynccnt 0x4" ::: "memory");
        __syncthreads();

        const char* bb = (ct & 1) ? (const char*)&bpanel[1][0]
                                  : (const char*)&bpanel[0][0];
        int c0 = ct * 16;
        v8f c = {};
#pragma unroll
        for (int ks = 0; ks < 8; ++ks) {
            Frag b;
            b.q[0] = *(const uint4*)(bb + ln * 512 + ks * 64 + k0 * 2);
            b.q[1] = *(const uint4*)(bb + ln * 512 + ks * 64 + k0 * 2 + 32);
            c = __builtin_amdgcn_wmma_f32_16x16x32_f16(
                    false, a[ks].h, false, b.h, (short)0, c, false, false);
        }
        int pj = pieces[c0 + ln];
        int j  = c0 + ln;
#pragma unroll
        for (int v = 0; v < 8; ++v) {
            int i = r0 + 8 * half + v;
            if (vld[v] && (pi[v] == pj) && (i != j)) {
                float x = c[v] * INV_T;
                acc += log1pf(__expf(nl[v] - x));
            }
        }
    }
    for (int off = 1; off < 32; off <<= 1) acc += __shfl_xor(acc, off, 32);
    if (lane == 0) atomicAdd(contrast_sum, acc);
}

__global__ void rank_loss_kernel(const float* __restrict__ logits,
                                 const float* __restrict__ la, const float* __restrict__ lb,
                                 float* __restrict__ rank_sum, unsigned int* __restrict__ rank_cnt) {
    __shared__ float rs[256];
    __shared__ unsigned int rc[256];
    int idx = blockIdx.x * blockDim.x + threadIdx.x;
    float sum = 0.0f; unsigned int cnt = 0u;
    if (idx < B_ROWS * NDIMS) {
        float ld = la[idx] - lb[idx];
        if (fabsf(ld) >= AMB_TH) {
            float t = (ld > 0.0f) ? (1.0f - 0.5f * LS) : (0.5f * LS);
            float x = logits[idx];
            sum = fmaxf(x, 0.0f) - x * t + log1pf(__expf(-fabsf(x)));
            cnt = 1u;
        }
    }
    rs[threadIdx.x] = sum; rc[threadIdx.x] = cnt;
    __syncthreads();
    for (int st = 128; st > 0; st >>= 1) {
        if (threadIdx.x < st) { rs[threadIdx.x] += rs[threadIdx.x + st]; rc[threadIdx.x] += rc[threadIdx.x + st]; }
        __syncthreads();
    }
    if (threadIdx.x == 0) { atomicAdd(rank_sum, rs[0]); atomicAdd(rank_cnt, rc[0]); }
}

__global__ void finalize_kernel(const int* __restrict__ pieces, const int* __restrict__ hist,
                                const float* __restrict__ scal_f, const unsigned int* __restrict__ scal_u,
                                float* __restrict__ out) {
    __shared__ long long red[256];
    long long pc = 0;
    for (int i = threadIdx.x; i < N_ROWS; i += 256) {
        int cnt = hist[pieces[i]];
        int pos = cnt - 1;
        if (pos >= 1 && (N_ROWS - 1 - pos) >= 1) pc += pos;
    }
    red[threadIdx.x] = pc;
    __syncthreads();
    for (int st = 128; st > 0; st >>= 1) {
        if (threadIdx.x < st) red[threadIdx.x] += red[threadIdx.x + st];
        __syncthreads();
    }
    if (threadIdx.x == 0) {
        long long count = red[0];
        float l_rank = (scal_u[2] > 0u) ? scal_f[1] / (float)scal_u[2] : 0.0f;
        float l_con  = (count > 0)      ? scal_f[0] / (float)count    : 0.0f;
        out[0] = l_rank + LAMBDA_C * l_con;
        out[1] = l_rank;
        out[2] = l_con;
    }
}

extern "C" void kernel_launch(void* const* d_in, const int* in_sizes, int n_in,
                              void* d_out, int out_size, void* d_ws, size_t ws_size,
                              hipStream_t stream) {
    const float* za     = (const float*)d_in[0];
    const float* zb     = (const float*)d_in[1];
    const float* logits = (const float*)d_in[2];
    const float* la     = (const float*)d_in[3];
    const float* lb     = (const float*)d_in[4];
    const int*   pa     = (const int*)d_in[5];
    const int*   pb     = (const int*)d_in[6];

    char* ws = (char*)d_ws;
    _Float16*     zh      = (_Float16*)ws;
    float*        neg_lse = (float*)(ws + 0x400000);
    int*          pieces  = (int*)(ws + 0x408000);
    int*          hist    = (int*)(ws + 0x410000);
    float*        scal_f  = (float*)(ws + 0x411000);
    unsigned int* scal_u  = (unsigned int*)(ws + 0x411000);

    init_ws_kernel<<<1, 256, 0, stream>>>((unsigned int*)hist);
    pieces_hist_kernel<<<N_ROWS / 256, 256, 0, stream>>>(pa, pb, pieces, hist);
    normalize_f16_kernel<<<N_ROWS, 256, 0, stream>>>(za, zb, zh);
    infonce_neglse_kernel<<<N_ROWS / 64, 128, 0, stream>>>(zh, pieces, neg_lse);
    infonce_pos_kernel<<<N_ROWS / 64, 128, 0, stream>>>(zh, pieces, neg_lse, hist, &scal_f[0]);
    rank_loss_kernel<<<(B_ROWS * NDIMS + 255) / 256, 256, 0, stream>>>(logits, la, lb, &scal_f[1], &scal_u[2]);
    finalize_kernel<<<1, 256, 0, stream>>>(pieces, hist, scal_f, scal_u, (float*)d_out);
}
